// ClusterForecasting_7344394076325
// MI455X (gfx1250) — matrix operand order repaired and checked by reading, not compiled, not run on hardware
//
#include <hip/hip_runtime.h>
#include <hip/hip_bf16.h>

// Problem constants
#define BB   4096   // batch
#define SS   64     // seq len
#define FF   8      // feature
#define DD   64     // d_model
#define KK   16     // knns
#define LROW 4096   // S*D = row length of x_re; also number of rows = BB

typedef __attribute__((ext_vector_type(16))) __bf16 v16bf;
typedef __attribute__((ext_vector_type(8)))  __bf16 v8bf;
typedef __attribute__((ext_vector_type(8)))  float  v8f;

// ---------------------------------------------------------------------------
// Kernel 1: x_enc = x @ W_emb + b_emb (store bf16), P = x_enc @ W_proj + b_proj
//           (store fp32 to out), and per-block partial sums of (P - x)^2.
// One thread per (b,s) row. 1024 blocks x 256 threads.
// ---------------------------------------------------------------------------
__global__ __launch_bounds__(256) void cf_embed_proj(
    const float* __restrict__ x, const float* __restrict__ We,
    const float* __restrict__ be, const float* __restrict__ Wp,
    const float* __restrict__ bp, unsigned short* __restrict__ xeu,
    float* __restrict__ outP, float* __restrict__ msePartial) {
  __shared__ float sWe[FF * DD];
  __shared__ float sWp[DD * FF];
  __shared__ float sBe[DD];
  __shared__ float sBp[FF];
  __shared__ float red[256];

  const int tid = threadIdx.x;
  for (int i = tid; i < FF * DD; i += 256) { sWe[i] = We[i]; sWp[i] = Wp[i]; }
  if (tid < DD) sBe[tid] = be[tid];
  if (tid < FF) sBp[tid] = bp[tid];
  __syncthreads();

  const int row = blockIdx.x * 256 + tid;          // [0, BB*SS)
  const float* xr = x + (size_t)row * FF;
  float xv[FF];
#pragma unroll
  for (int f = 0; f < FF; ++f) xv[f] = xr[f];

  float p[FF];
#pragma unroll
  for (int f = 0; f < FF; ++f) p[f] = sBp[f];

  __bf16* xe = reinterpret_cast<__bf16*>(xeu) + (size_t)row * DD;
#pragma unroll 2
  for (int d0 = 0; d0 < DD; d0 += 8) {
    float ev[8];
#pragma unroll
    for (int j = 0; j < 8; ++j) {
      const int d = d0 + j;
      float e = sBe[d];
#pragma unroll
      for (int f = 0; f < FF; ++f) e += xv[f] * sWe[f * DD + d];
      ev[j] = e;
#pragma unroll
      for (int fo = 0; fo < FF; ++fo) p[fo] += e * sWp[d * FF + fo];
    }
    v8bf vb;
#pragma unroll
    for (int j = 0; j < 8; ++j) vb[j] = (__bf16)ev[j];
    *reinterpret_cast<v8bf*>(xe + d0) = vb;        // 16B aligned store
  }

  float local = 0.f;
  float* po = outP + (size_t)row * FF;
#pragma unroll
  for (int fo = 0; fo < FF; ++fo) {
    po[fo] = p[fo];
    const float dd = p[fo] - xv[fo];
    local += dd * dd;
  }

  red[tid] = local;
  __syncthreads();
  for (int off = 128; off > 0; off >>= 1) {
    if (tid < off) red[tid] += red[tid + off];
    __syncthreads();
  }
  if (tid == 0) msePartial[blockIdx.x] = red[0];
}

// ---------------------------------------------------------------------------
// Kernel 2: attn = (x_re @ x_re^T) / 64 with diag = -inf.  bf16 WMMA.
// 128x128 tile per 256-thread workgroup (8 waves, each 32x64 = 2x4 C tiles).
// Double-buffered LDS; tiles staged with GLOBAL_LOAD_ASYNC_TO_LDS_B128
// (ASYNCcnt) so the next tile's global traffic overlaps current WMMAs.
// ---------------------------------------------------------------------------
__global__ __launch_bounds__(256) void cf_gram_wmma(
    const unsigned short* __restrict__ xeu, float* __restrict__ attn) {
  // padded LDS stride 40 halves (80 B) to spread banks; 2 buffers each side
  __shared__ __align__(16) unsigned short As[2][128 * 40];
  __shared__ __align__(16) unsigned short Bs[2][128 * 40];

  const int tid  = threadIdx.x;
  const int lane = tid & 31;
  const int wave = tid >> 5;
  const int bm = blockIdx.x * 128;
  const int bn = blockIdx.y * 128;
  const int wM = (wave >> 1) * 32;   // 4 wave-rows of 32
  const int wN = (wave & 1) * 64;    // 2 wave-cols of 64

  // This thread's staging assignment: 4 x b128 per K-step (2 per tile side).
  // i = tid + j*256 -> row i>>2, 8-half segment i&3.
  const int r0 = tid >> 2, seg0 = tid & 3;
  const int r1 = (tid + 256) >> 2, seg1 = (tid + 256) & 3;

  auto stage = [&](int buf, int kk) {
    const unsigned short* ga0 = xeu + (size_t)(bm + r0) * LROW + kk + seg0 * 8;
    const unsigned short* gb0 = xeu + (size_t)(bn + r0) * LROW + kk + seg0 * 8;
    const unsigned short* ga1 = xeu + (size_t)(bm + r1) * LROW + kk + seg1 * 8;
    const unsigned short* gb1 = xeu + (size_t)(bn + r1) * LROW + kk + seg1 * 8;
    // Flat address of LDS = {aperture, lds_offset}; low 32 bits = LDS offset.
    const unsigned la0 = (unsigned)(size_t)&As[buf][r0 * 40 + seg0 * 8];
    const unsigned lb0 = (unsigned)(size_t)&Bs[buf][r0 * 40 + seg0 * 8];
    const unsigned la1 = (unsigned)(size_t)&As[buf][r1 * 40 + seg1 * 8];
    const unsigned lb1 = (unsigned)(size_t)&Bs[buf][r1 * 40 + seg1 * 8];
    asm volatile("global_load_async_to_lds_b128 %0, %1, off"
                 :: "v"(la0), "v"(ga0) : "memory");
    asm volatile("global_load_async_to_lds_b128 %0, %1, off"
                 :: "v"(lb0), "v"(gb0) : "memory");
    asm volatile("global_load_async_to_lds_b128 %0, %1, off"
                 :: "v"(la1), "v"(ga1) : "memory");
    asm volatile("global_load_async_to_lds_b128 %0, %1, off"
                 :: "v"(lb1), "v"(gb1) : "memory");
  };

  v8f acc[2][4];
#pragma unroll
  for (int m = 0; m < 2; ++m)
#pragma unroll
    for (int n = 0; n < 4; ++n) acc[m][n] = (v8f)(0.f);

  const int mrow = lane & 15;
  const int hiA  = (lane >= 16) ? 8 : 0;    // A: K base offset within row
  const int hiB  = (lane >= 16) ? 8 : 0;    // B: K base offset within row

  // Preload first tile pair.
  stage(0, 0);
  asm volatile("s_wait_asynccnt 0x0" ::: "memory");
  __syncthreads();

  int cur = 0;
  for (int kk = 0; kk < LROW; kk += 32) {
    const int nxt = cur ^ 1;
    if (kk + 32 < LROW) stage(nxt, kk + 32);   // overlap with compute below

    // A fragments: 16x32 bf16, lane L<16 -> M=L, K in {0..7,16..23}; L>=16 -> +8
    v16bf a[2];
#pragma unroll
    for (int m = 0; m < 2; ++m) {
      const __bf16* pa =
          reinterpret_cast<const __bf16*>(&As[cur][(wM + m * 16 + mrow) * 40]);
      const v8bf lo = *reinterpret_cast<const v8bf*>(pa + hiA);
      const v8bf hi = *reinterpret_cast<const v8bf*>(pa + hiA + 16);
      a[m] = __builtin_shufflevector(lo, hi, 0, 1, 2, 3, 4, 5, 6, 7, 8, 9, 10,
                                     11, 12, 13, 14, 15);
    }
    // B fragments: 32x16 bf16, lane = N column; K in {0..7,16..23} (+8 for hi lanes).
    // B[k][n] = x_re[bn + n][k] -> read row (bn+n) of x_re.
    v16bf b[4];
#pragma unroll
    for (int n = 0; n < 4; ++n) {
      const __bf16* pb =
          reinterpret_cast<const __bf16*>(&Bs[cur][(wN + n * 16 + mrow) * 40]);
      const v8bf lo = *reinterpret_cast<const v8bf*>(pb + hiB);
      const v8bf hi = *reinterpret_cast<const v8bf*>(pb + hiB + 16);
      b[n] = __builtin_shufflevector(lo, hi, 0, 1, 2, 3, 4, 5, 6, 7, 8, 9, 10,
                                     11, 12, 13, 14, 15);
    }
#pragma unroll
    for (int m = 0; m < 2; ++m)
#pragma unroll
      for (int n = 0; n < 4; ++n)
        acc[m][n] = __builtin_amdgcn_wmma_f32_16x16x32_bf16(
            false, a[m], false, b[n], (short)0, acc[m][n], false, false);

    // Next buffer must be fully landed (this wave's async ops) and all waves
    // must be past their reads of `cur` before it is reused.
    asm volatile("s_wait_asynccnt 0x0" ::: "memory");
    __syncthreads();
    cur = nxt;
  }

  // Epilogue: C layout — lane&15 = N, VGPR v -> M = v + 8*(lane>=16)
  const float scale = 1.0f / 64.0f;   // 1/sqrt(D*S)
  const int mAdd = (lane >= 16) ? 8 : 0;
#pragma unroll
  for (int m = 0; m < 2; ++m) {
#pragma unroll
    for (int n = 0; n < 4; ++n) {
#pragma unroll
      for (int v = 0; v < 8; ++v) {
        const int gm = bm + wM + m * 16 + v + mAdd;
        const int gn = bn + wN + n * 16 + (lane & 15);
        const float val = acc[m][n][v] * scale;
        attn[(size_t)gm * LROW + gn] = (gm == gn) ? -__builtin_inff() : val;
      }
    }
  }
}

// ---------------------------------------------------------------------------
// Kernel 3: top-16 per row of attn (softmax is monotonic -> logits suffice).
// Ties go to the lower index, matching jax.lax.top_k. One block per row.
// ---------------------------------------------------------------------------
__global__ __launch_bounds__(256) void cf_topk16(const float* __restrict__ attn,
                                                 int* __restrict__ idxOut) {
  __shared__ float sv[256];
  __shared__ int si[256];
  __shared__ int sWin;
  const int row = blockIdx.x;
  const int tid = threadIdx.x;

  float vals[16];
#pragma unroll
  for (int i = 0; i < 16; ++i)
    vals[i] = attn[(size_t)row * LROW + tid + i * 256];

  for (int it = 0; it < KK; ++it) {
    float bv = vals[0];
    int bi = tid;
#pragma unroll
    for (int i = 1; i < 16; ++i) {
      const int c = tid + i * 256;
      const float v = vals[i];
      if (v > bv || (v == bv && c < bi)) { bv = v; bi = c; }
    }
    sv[tid] = bv;
    si[tid] = bi;
    __syncthreads();
    for (int off = 128; off > 0; off >>= 1) {
      if (tid < off) {
        const float ov = sv[tid + off];
        const int oi = si[tid + off];
        if (ov > sv[tid] || (ov == sv[tid] && oi < si[tid])) {
          sv[tid] = ov;
          si[tid] = oi;
        }
      }
      __syncthreads();
    }
    if (tid == 0) {
      sWin = si[0];
      idxOut[row * KK + it] = si[0];
    }
    __syncthreads();
    const int win = sWin;
    if ((win & 255) == tid) vals[win >> 8] = -__builtin_inff();
    __syncthreads();
  }
}

// ---------------------------------------------------------------------------
// Kernel 4: diff_1 / diff_2 partial sums over gathered = P[top_idx].
// One block per batch element b. P rows are S*F = 512 floats.
// ---------------------------------------------------------------------------
__global__ __launch_bounds__(256) void cf_diffs(const float* __restrict__ P,
                                                const int* __restrict__ idx,
                                                float* __restrict__ d1Partial,
                                                float* __restrict__ d2Partial) {
  __shared__ int sIdx[KK];
  __shared__ float red[256];
  const int b = blockIdx.x;
  const int tid = threadIdx.x;
  if (tid < KK) sIdx[tid] = idx[b * KK + tid];
  __syncthreads();

  float a1 = 0.f, a2 = 0.f;
  for (int k = 0; k < KK; ++k) {
    const float* Pk = P + (size_t)sIdx[k] * (SS * FF);
    const float* Pk1 =
        (k < KK - 1) ? (P + (size_t)sIdx[k + 1] * (SS * FF)) : Pk;
    for (int t = tid; t < SS * FF; t += 256) {
      const float v = Pk[t];
      if (k < KK - 1) {
        const float d = Pk1[t] - v;
        a1 += d * d;
      }
      if (t < (SS - 1) * FF) {                 // diff along s: t vs t+F
        const float d = Pk[t + FF] - v;
        a2 += d * d;
      }
    }
  }

  red[tid] = a1;
  __syncthreads();
  for (int off = 128; off > 0; off >>= 1) {
    if (tid < off) red[tid] += red[tid + off];
    __syncthreads();
  }
  if (tid == 0) d1Partial[b] = red[0];
  __syncthreads();
  red[tid] = a2;
  __syncthreads();
  for (int off = 128; off > 0; off >>= 1) {
    if (tid < off) red[tid] += red[tid + off];
    __syncthreads();
  }
  if (tid == 0) d2Partial[b] = red[0];
}

// ---------------------------------------------------------------------------
// Kernel 5: final deterministic reduction -> loss scalar.
// ---------------------------------------------------------------------------
__global__ __launch_bounds__(256) void cf_finalize(
    const float* __restrict__ msePartial, const float* __restrict__ d1P,
    const float* __restrict__ d2P, float* __restrict__ out) {
  __shared__ float r0[256], r1[256], r2[256];
  const int tid = threadIdx.x;
  float s0 = 0.f, s1 = 0.f, s2 = 0.f;
  for (int i = tid; i < 1024; i += 256) s0 += msePartial[i];
  for (int i = tid; i < 4096; i += 256) { s1 += d1P[i]; s2 += d2P[i]; }
  r0[tid] = s0; r1[tid] = s1; r2[tid] = s2;
  __syncthreads();
  for (int off = 128; off > 0; off >>= 1) {
    if (tid < off) {
      r0[tid] += r0[tid + off];
      r1[tid] += r1[tid + off];
      r2[tid] += r2[tid + off];
    }
    __syncthreads();
  }
  if (tid == 0) {
    const float mse = r0[0] / 2097152.0f;          // B*S*F
    const float d1  = r1[0] / 31457280.0f;         // B*(K-1)*S*F
    const float d2  = r2[0] / 33030144.0f;         // B*K*(S-1)*F
    out[0] = mse + d1 + d2;
  }
}

// ---------------------------------------------------------------------------
extern "C" void kernel_launch(void* const* d_in, const int* in_sizes, int n_in,
                              void* d_out, int out_size, void* d_ws,
                              size_t ws_size, hipStream_t stream) {
  (void)in_sizes; (void)n_in; (void)out_size; (void)ws_size;
  const float* x  = (const float*)d_in[0];
  const float* We = (const float*)d_in[1];
  const float* be = (const float*)d_in[2];
  const float* Wp = (const float*)d_in[3];
  const float* bp = (const float*)d_in[4];

  float* out  = (float*)d_out;          // out[0] = loss, out+1 = x_rec_proj
  float* P    = out + 1;

  char* ws = (char*)d_ws;
  float* msePartial = (float*)ws;                                   // 1024 f
  float* d1P        = (float*)(ws + 4096);                          // 4096 f
  float* d2P        = (float*)(ws + 4096 + 16384);                  // 4096 f
  int*   topIdx     = (int*)(ws + 4096 + 16384 + 16384);            // 65536 i
  unsigned short* xe = (unsigned short*)(ws + (1 << 20));           // 32 MB
  float* attn = (float*)(ws + (1 << 20) + (size_t)LROW * LROW * 2); // 64 MB

  cf_embed_proj<<<dim3((BB * SS) / 256), dim3(256), 0, stream>>>(
      x, We, be, Wp, bp, xe, P, msePartial);

  cf_gram_wmma<<<dim3(LROW / 128, LROW / 128), dim3(256), 0, stream>>>(xe,
                                                                       attn);

  cf_topk16<<<dim3(BB), dim3(256), 0, stream>>>(attn, topIdx);

  cf_diffs<<<dim3(BB), dim3(256), 0, stream>>>(P, topIdx, d1P, d2P);

  cf_finalize<<<dim3(1), dim3(256), 0, stream>>>(msePartial, d1P, d2P, out);
}
